// PNNLayer_40080634807072
// MI455X (gfx1250) — matrix operand order repaired and checked by reading, not compiled, not run on hardware
//
#include <hip/hip_runtime.h>

typedef __attribute__((ext_vector_type(16))) __bf16 v16bf;
typedef __attribute__((ext_vector_type(8)))  __bf16 v8bf;
typedef __attribute__((ext_vector_type(8)))  float  v8f;
typedef __attribute__((__vector_size__(4 * sizeof(int)))) int v4i;

#define M_DIM 8192
#define N_DIM 4096
#define K_IN  4096
#define K_AUG 4098   // n_in + 2 (ones col, zeros col)
#define K_PAD 4128   // K_AUG padded up to multiple of 32
#define K2    8256   // 2*K_PAD : [a_aug | inv_a] fused-K GEMM
#define BM 128
#define BN 128
#define BK 32
#define LDS_STRIDE 40   // BK + 8 bf16 pad (80 B rows, 16B-aligned slots)
#define NT (K2 / BK)    // 258 K-steps

#define AS1 __attribute__((address_space(1)))
#define AS3 __attribute__((address_space(3)))

#if defined(__gfx1250__) && __has_builtin(__builtin_amdgcn_global_load_async_to_lds_b128)
#define USE_ASYNC 1
#else
#define USE_ASYNC 0
#endif

__device__ __forceinline__ void wait_async0() {
#if defined(__gfx1250__)
#if __has_builtin(__builtin_amdgcn_s_wait_asynccnt)
    __builtin_amdgcn_s_wait_asynccnt(0);
#else
    asm volatile("s_wait_asynccnt 0" ::: "memory");
#endif
#endif
}

// Branchless tanh: 1 - 2*rcp(exp2(2*log2e*x)+1); saturates to +/-1 at the
// extremes through v_exp_f32 / v_rcp_f32 semantics. ~5 VALU + 1 TRANS.
__device__ __forceinline__ float fast_tanh(float x) {
    float t = __builtin_amdgcn_exp2f(x * 2.8853900817779268f);
    return 1.0f - 2.0f * __builtin_amdgcn_rcpf(t + 1.0f);
}
__device__ __forceinline__ float inv_fn(float x) {
    return 0.104f - 0.899f * fast_tanh((x + 0.056f) * 3.858f);
}
__device__ __forceinline__ float act_fn(float z) {
    return 0.134f + 0.962f * fast_tanh((z - 0.183f) * 24.1f);
}

// ---------------------------------------------------------------------------
// Prep A2[j][c] (bf16):  c < K_PAD  -> a_aug  (a_prev | 1 | 0 | pad0)
//                        c >= K_PAD -> inv(a_aug) (pad cols -> 0)
// ---------------------------------------------------------------------------
__global__ void prep_A(const float* __restrict__ a_prev, __bf16* __restrict__ A2) {
    const int j = blockIdx.y;
    const int c = blockIdx.x * blockDim.x + threadIdx.x;
    if (c >= K2) return;
    float v;
    if (c < K_PAD) {
        if (c < K_IN)        v = a_prev[(size_t)j * K_IN + c];
        else if (c == K_IN)  v = 1.0f;
        else                 v = 0.0f;               // zeros col + pad
    } else {
        const int c2 = c - K_PAD;
        if (c2 < K_AUG) {
            float a = (c2 < K_IN) ? a_prev[(size_t)j * K_IN + c2]
                                  : (c2 == K_IN ? 1.0f : 0.0f);
            v = inv_fn(a);
        } else {
            v = 0.0f;                                // pad (B2 pad is 0 too)
        }
    }
    A2[(size_t)j * K2 + c] = (__bf16)v;
}

// ---------------------------------------------------------------------------
// Prep B2[i][k] (bf16): th = st(theta*aging); g=|th|; W=g/rowsum(g)
//   first K_PAD cols  : W where th>=0 else 0
//   second K_PAD cols : W where th<0  else 0
// ---------------------------------------------------------------------------
__global__ void prep_B(const float* __restrict__ theta,
                       const float* __restrict__ aging,
                       __bf16* __restrict__ B2) {
    __shared__ float red[256];
    const int i   = blockIdx.x;
    const int tid = threadIdx.x;
    const float* tr = theta + (size_t)i * K_AUG;
    const float* ar = aging + (size_t)i * K_AUG;

    float thv[17];
    float s = 0.0f;
#pragma unroll
    for (int it = 0; it < 17; ++it) {
        const int k = tid + it * 256;
        float th = 0.0f;
        if (k < K_AUG) {
            th = tr[k] * ar[k];
            if (fabsf(th) < 0.01f) th = 0.0f;        // straight-through fwd
            s += fabsf(th);
        }
        thv[it] = th;
    }
    red[tid] = s;
    __syncthreads();
    for (int off = 128; off > 0; off >>= 1) {
        if (tid < off) red[tid] += red[tid + off];
        __syncthreads();
    }
    const float rinv = 1.0f / red[0];

#pragma unroll
    for (int it = 0; it < 17; ++it) {
        const int k = tid + it * 256;
        if (k < K_PAD) {
            float bp = 0.0f, bn = 0.0f;
            if (k < K_AUG) {
                const float th = thv[it];
                const float w  = fabsf(th) * rinv;
                if (th >= 0.0f) bp = w; else bn = w;
            }
            B2[(size_t)i * K2 + k]         = (__bf16)bp;
            B2[(size_t)i * K2 + K_PAD + k] = (__bf16)bn;
        }
    }
}

// ---------------------------------------------------------------------------
// Fused GEMM + activation, double-buffered LDS with async global->LDS copies.
// 128x128 block tile, 8 wave32 waves, each wave = 64x32 = 4x2 WMMA tiles,
// v_wmma_f32_16x16x32_bf16 per K-step of 32.
// ---------------------------------------------------------------------------
__global__ void __launch_bounds__(256)
gemm_act(const __bf16* __restrict__ A2, const __bf16* __restrict__ B2,
         float* __restrict__ out) {
    __shared__ __bf16 As[2][BM * LDS_STRIDE];
    __shared__ __bf16 Bs[2][BN * LDS_STRIDE];

    const int tid   = threadIdx.x;
    const int lane  = tid & 31;
    const int wave  = tid >> 5;
    const int wm    = wave >> 2;      // 0..1  (M sub-tile of 64)
    const int wn    = wave & 3;       // 0..3  (N sub-tile of 32)
    const int ln16  = lane & 15;
    const int lhalf = lane >> 4;      // 0 or 1

    const int m0 = blockIdx.y * BM;
    const int n0 = blockIdx.x * BN;

    // this thread's two 16B chunks of the 128x32 tile
    const int row0 = tid >> 2;                 // chunk 0
    const int col0 = (tid & 3) << 3;
    const int row1 = (tid + 256) >> 2;         // chunk 1
    const int col1 = ((tid + 256) & 3) << 3;

    v8f acc[4][2];
#pragma unroll
    for (int a = 0; a < 4; ++a)
#pragma unroll
        for (int b = 0; b < 2; ++b) {
            v8f z = {0.f, 0.f, 0.f, 0.f, 0.f, 0.f, 0.f, 0.f};
            acc[a][b] = z;
        }

    auto issue_tile = [&](int k0, int buf) {
#if USE_ASYNC
        __builtin_amdgcn_global_load_async_to_lds_b128(
            (AS1 v4i*)(A2 + (size_t)(m0 + row0) * K2 + k0 + col0),
            (AS3 v4i*)(&As[buf][row0 * LDS_STRIDE + col0]), 0, 0);
        __builtin_amdgcn_global_load_async_to_lds_b128(
            (AS1 v4i*)(A2 + (size_t)(m0 + row1) * K2 + k0 + col1),
            (AS3 v4i*)(&As[buf][row1 * LDS_STRIDE + col1]), 0, 0);
        __builtin_amdgcn_global_load_async_to_lds_b128(
            (AS1 v4i*)(B2 + (size_t)(n0 + row0) * K2 + k0 + col0),
            (AS3 v4i*)(&Bs[buf][row0 * LDS_STRIDE + col0]), 0, 0);
        __builtin_amdgcn_global_load_async_to_lds_b128(
            (AS1 v4i*)(B2 + (size_t)(n0 + row1) * K2 + k0 + col1),
            (AS3 v4i*)(&Bs[buf][row1 * LDS_STRIDE + col1]), 0, 0);
#else
        *(uint4*)(&As[buf][row0 * LDS_STRIDE + col0]) =
            *(const uint4*)(A2 + (size_t)(m0 + row0) * K2 + k0 + col0);
        *(uint4*)(&As[buf][row1 * LDS_STRIDE + col1]) =
            *(const uint4*)(A2 + (size_t)(m0 + row1) * K2 + k0 + col1);
        *(uint4*)(&Bs[buf][row0 * LDS_STRIDE + col0]) =
            *(const uint4*)(B2 + (size_t)(n0 + row0) * K2 + k0 + col0);
        *(uint4*)(&Bs[buf][row1 * LDS_STRIDE + col1]) =
            *(const uint4*)(B2 + (size_t)(n0 + row1) * K2 + k0 + col1);
#endif
    };

    issue_tile(0, 0);

    for (int kt = 0; kt < NT; ++kt) {
        const int buf = kt & 1;
#if USE_ASYNC
        wait_async0();                 // loads for tile kt have landed in LDS
#endif
        __syncthreads();               // all waves see tile kt; tile kt-1 reads done
        if (kt + 1 < NT)               // overlap: fetch kt+1 into the other buffer
            issue_tile((kt + 1) * BK, buf ^ 1);

        // A fragment: lanes 0-15: M=lane, K {0..7,16..23}; lanes 16-31: K {8..15,24..31}
        v16bf afrag[4];
#pragma unroll
        for (int mt = 0; mt < 4; ++mt) {
            const int r = wm * 64 + mt * 16 + ln16;
            v8bf lo = *(const v8bf*)(&As[buf][r * LDS_STRIDE + lhalf * 8]);
            v8bf hi = *(const v8bf*)(&As[buf][r * LDS_STRIDE + lhalf * 8 + 16]);
            afrag[mt] = __builtin_shufflevector(lo, hi,
                0, 1, 2, 3, 4, 5, 6, 7, 8, 9, 10, 11, 12, 13, 14, 15);
        }
        // B fragment: N=lane%16; lanes 0-15: K {0..7,16..23}; lanes 16-31: K {8..15,24..31}
        v16bf bfrag[2];
#pragma unroll
        for (int nt = 0; nt < 2; ++nt) {
            const int r = wn * 32 + nt * 16 + ln16;
            v8bf lo = *(const v8bf*)(&Bs[buf][r * LDS_STRIDE + lhalf * 8]);
            v8bf hi = *(const v8bf*)(&Bs[buf][r * LDS_STRIDE + lhalf * 8 + 16]);
            bfrag[nt] = __builtin_shufflevector(lo, hi,
                0, 1, 2, 3, 4, 5, 6, 7, 8, 9, 10, 11, 12, 13, 14, 15);
        }

#pragma unroll
        for (int mt = 0; mt < 4; ++mt)
#pragma unroll
            for (int nt = 0; nt < 2; ++nt)
                acc[mt][nt] = __builtin_amdgcn_wmma_f32_16x16x32_bf16(
                    false, afrag[mt], false, bfrag[nt],
                    (short)0, acc[mt][nt], false, false);
    }

    // Epilogue: C layout -> VGPR r holds M = 8*lhalf + r, N = lane%16
#pragma unroll
    for (int mt = 0; mt < 4; ++mt) {
#pragma unroll
        for (int nt = 0; nt < 2; ++nt) {
            const int col = n0 + wn * 32 + nt * 16 + ln16;
#pragma unroll
            for (int r = 0; r < 8; ++r) {
                const int row = m0 + wm * 64 + mt * 16 + lhalf * 8 + r;
                out[(size_t)row * N_DIM + col] = act_fn(acc[mt][nt][r]);
            }
        }
    }
}

extern "C" void kernel_launch(void* const* d_in, const int* in_sizes, int n_in,
                              void* d_out, int out_size, void* d_ws, size_t ws_size,
                              hipStream_t stream) {
    (void)in_sizes; (void)n_in; (void)out_size; (void)ws_size;
    const float* a_prev = (const float*)d_in[0];
    const float* theta  = (const float*)d_in[1];
    const float* aging  = (const float*)d_in[2];
    float* out = (float*)d_out;

    __bf16* A2 = (__bf16*)d_ws;                    // [M_DIM x K2] bf16
    __bf16* B2 = A2 + (size_t)M_DIM * K2;          // [N_DIM x K2] bf16

    dim3 gA((K2 + 255) / 256, M_DIM);
    prep_A<<<gA, 256, 0, stream>>>(a_prev, A2);
    prep_B<<<N_DIM, 256, 0, stream>>>(theta, aging, B2);

    dim3 gG(N_DIM / BN, M_DIM / BM);
    gemm_act<<<gG, 256, 0, stream>>>(A2, B2, out);
}
